// CESoftmaxAttention_32590211842337
// MI455X (gfx1250) — matrix operand order repaired and checked by reading, not compiled, not run on hardware
//
#include <hip/hip_runtime.h>
#include <hip/hip_bf16.h>
#include <math.h>

typedef __bf16 bf16_t;
typedef __attribute__((ext_vector_type(16))) __bf16 v16bf;
typedef __attribute__((ext_vector_type(8)))  __bf16 v8bf;
typedef __attribute__((ext_vector_type(8)))  float  v8f;

#define D_MODEL 1024
#define NTOK    2048
#define BATCH   2
#define NHEAD   16
#define DHEAD   64
#define MTOT    (BATCH*NTOK)   /* 4096 tokens total */

#define CL 0.085625f   /* 0.685 / sqrt(64) */
#define CG 0.1575f     /* 0.315 / 2       */

// ---------------------------------------------------------------------------
// WMMA helpers (CDNA5 gfx1250, wave32): D = A(16x32 bf16) * B(32x16 bf16) + C
// ---------------------------------------------------------------------------
__device__ __forceinline__ v8f wmma_bf16(v16bf a, v16bf b, v8f c) {
  return __builtin_amdgcn_wmma_f32_16x16x32_bf16(
      /*neg_a=*/false, a, /*neg_b=*/false, b,
      /*c_mod=*/(short)0, c, /*reuse_a=*/false, /*reuse_b=*/false);
}

// A fragment: 16x32, rows contiguous in memory with leading dim `ld`.
// lane<16: row=lane,   K = {0..7, 16..23}
// lane>=16: row=lane-16, K = {8..15, 24..31}
__device__ __forceinline__ v16bf load_a_frag(const bf16_t* base, int ld, int lane) {
  const int half = lane >> 4;
  const int r    = lane & 15;
  const bf16_t* p = base + (size_t)r * ld + half * 8;
  v8bf lo = *(const v8bf*)(p);        // K = half*8 + 0..7
  v8bf hi = *(const v8bf*)(p + 16);   // K = 16 + half*8 + 0..7
  v16bf a;
#pragma unroll
  for (int i = 0; i < 8; ++i) { a[i] = lo[i]; a[8 + i] = hi[i]; }
  return a;
}

// B fragment: 32x16 (KxN), columns contiguous in K with column stride `ldc`.
// lane<16: col=lane, K = {0..7, 16..23} ; lane>=16: col=lane-16, K = {8..15, 24..31}
__device__ __forceinline__ v16bf load_b_frag(const bf16_t* base, int ldc, int lane) {
  const int half = lane >> 4;
  const int n    = lane & 15;
  const bf16_t* p = base + (size_t)n * ldc + half * 8;
  v8bf lo = *(const v8bf*)(p);
  v8bf hi = *(const v8bf*)(p + 16);
  v16bf b;
#pragma unroll
  for (int i = 0; i < 8; ++i) { b[i] = lo[i]; b[8 + i] = hi[i]; }
  return b;
}

// ---------------------------------------------------------------------------
// Precision conversion / layout kernels
// ---------------------------------------------------------------------------
__global__ void cvt_f32_bf16(const float* __restrict__ in, bf16_t* __restrict__ out, int count) {
  int i = blockIdx.x * blockDim.x + threadIdx.x;
  int stride = gridDim.x * blockDim.x;
  for (; i < count; i += stride) out[i] = (bf16_t)in[i];
}

// W[k][n] f32 -> Wt[n][k] bf16 (coalesced read over n)
__global__ void transpose_w_bf16(const float* __restrict__ W, bf16_t* __restrict__ Wt) {
  int idx = blockIdx.x * blockDim.x + threadIdx.x;
  int k = idx >> 10, nn = idx & 1023;
  Wt[(size_t)nn * D_MODEL + k] = (bf16_t)W[idx];
}

// sq[(b*16+h)*2048 + n] = sum_d K[b,n,h*64+d]^2
__global__ void k_sqnorm(const bf16_t* __restrict__ K, float* __restrict__ sq) {
  int idx = blockIdx.x * blockDim.x + threadIdx.x; // B*H*N = 65536
  int nl = idx & (NTOK - 1);
  int bh = idx >> 11;
  int b = bh >> 4, h = bh & 15;
  const bf16_t* p = K + ((size_t)(b * NTOK + nl)) * D_MODEL + h * DHEAD;
  float s = 0.f;
#pragma unroll
  for (int v = 0; v < 8; ++v) {
    v8bf x = *(const v8bf*)(p + v * 8);
#pragma unroll
    for (int i = 0; i < 8; ++i) { float f = (float)x[i]; s += f * f; }
  }
  sq[idx] = s;
}

// ---------------------------------------------------------------------------
// WMMA GEMM:  C[4096 x 1024] = A[4096 x 1024] * Bt^T   (Bt is [n][k] bf16)
// MODE 0: store bf16 token-major   (Q, K)
// MODE 1: store bf16 transposed    (V -> Vt[b,h,d,token])
// MODE 2: store f32  token-major   (final output)
// block = 128 threads (4 waves); block tile 64x64, each wave 16x64.
// ---------------------------------------------------------------------------
template <int MODE>
__global__ void gemm_proj(const bf16_t* __restrict__ A,
                          const bf16_t* __restrict__ Bt,
                          void* __restrict__ Cout) {
  const int lane = threadIdx.x & 31;
  const int wave = threadIdx.x >> 5;
  const int row0 = blockIdx.x * 64 + wave * 16;
  const int col0 = blockIdx.y * 64;
  v8f acc[4] = {};
  for (int k0 = 0; k0 < D_MODEL; k0 += 32) {
    v16bf a = load_a_frag(A + (size_t)row0 * D_MODEL + k0, D_MODEL, lane);
#pragma unroll
    for (int t = 0; t < 4; ++t) {
      v16bf b = load_b_frag(Bt + (size_t)(col0 + t * 16) * D_MODEL + k0, D_MODEL, lane);
      acc[t] = wmma_bf16(a, b, acc[t]);
    }
  }
  const int n = lane & 15, half = lane >> 4;
  if (MODE == 0) {
    bf16_t* C = (bf16_t*)Cout;
#pragma unroll
    for (int t = 0; t < 4; ++t)
#pragma unroll
      for (int i = 0; i < 8; ++i) {
        int row = row0 + half * 8 + i;
        C[(size_t)row * D_MODEL + col0 + t * 16 + n] = (bf16_t)acc[t][i];
      }
  } else if (MODE == 1) {
    bf16_t* Vt = (bf16_t*)Cout;
    int tok0 = row0 + half * 8;
    int b = tok0 >> 11, nn = tok0 & (NTOK - 1);
#pragma unroll
    for (int t = 0; t < 4; ++t) {
      int dcol = col0 + t * 16 + n; // global feature = h*64 + d
      v8bf pk;
#pragma unroll
      for (int i = 0; i < 8; ++i) pk[i] = (bf16_t)acc[t][i];
      *(v8bf*)(Vt + ((size_t)(b * D_MODEL + dcol)) * NTOK + nn) = pk;
    }
  } else {
    float* C = (float*)Cout;
#pragma unroll
    for (int t = 0; t < 4; ++t)
#pragma unroll
      for (int i = 0; i < 8; ++i) {
        int row = row0 + half * 8 + i;
        C[(size_t)row * D_MODEL + col0 + t * 16 + n] = acc[t][i];
      }
  }
}

// ---------------------------------------------------------------------------
// Fused flash-attention with gravity kernel.
// One wave per (b, h, 16-query tile); loop over 32-key tiles, processed as
// two sequential 16-key sub-tiles so only one S and one G accumulator (and
// one B-frag pair) are live at a time -> no scratch spills.
//   S = 0.685/8 * (Q K^T) - 0.315/2 * max(|k_i|^2+|k_j|^2-2 K K^T, 0)
// Online softmax; P routed through LDS (transpose to A-frag); P*V via WMMA.
// __launch_bounds__(32, 1): single wave per block, allow max VGPR budget.
// ---------------------------------------------------------------------------
__global__ void __launch_bounds__(32, 1)
attn_kernel(const bf16_t* __restrict__ Q,
            const bf16_t* __restrict__ K,
            const bf16_t* __restrict__ Vt,
            const float*  __restrict__ sq,
            bf16_t* __restrict__ O) {
  __shared__ bf16_t pLds[16 * 32];
  const int lane = threadIdx.x & 31;
  const int qt = blockIdx.x, h = blockIdx.y, b = blockIdx.z;
  const int qrow0 = qt * 16;
  const int half = lane >> 4, n = lane & 15;

  const size_t qkBase = ((size_t)(b * NTOK + qrow0)) * D_MODEL + h * DHEAD;
  v16bf aq0 = load_a_frag(Q + qkBase,      D_MODEL, lane); // d = 0..31
  v16bf aq1 = load_a_frag(Q + qkBase + 32, D_MODEL, lane); // d = 32..63
  v16bf ak0 = load_a_frag(K + qkBase,      D_MODEL, lane);
  v16bf ak1 = load_a_frag(K + qkBase + 32, D_MODEL, lane);

  const float* sqBH = sq + ((size_t)(b * NHEAD + h)) * NTOK;
  float sqi[8];
#pragma unroll
  for (int i = 0; i < 8; ++i) sqi[i] = sqBH[qrow0 + half * 8 + i];

  v8f o[4] = {};
  float mrow[8], lrow[8];
#pragma unroll
  for (int i = 0; i < 8; ++i) { mrow[i] = -3.0e38f; lrow[i] = 0.f; }

  for (int j0 = 0; j0 < NTOK; j0 += 32) {
    const size_t kBase = ((size_t)(b * NTOK + j0)) * D_MODEL + h * DHEAD;

    // Two 16-key sub-tiles, sequential to keep register pressure low.
    float sv[2][8];
#pragma unroll
    for (int t2 = 0; t2 < 2; ++t2) {
      const bf16_t* kb = K + kBase + (size_t)(t2 * 16) * D_MODEL;
      v16bf bkl = load_b_frag(kb,      D_MODEL, lane); // d 0..31
      v16bf bkh = load_b_frag(kb + 32, D_MODEL, lane); // d 32..63
      v8f s = {}, g = {};
      s = wmma_bf16(aq0, bkl, s);  s = wmma_bf16(aq1, bkh, s);
      g = wmma_bf16(ak0, bkl, g);  g = wmma_bf16(ak1, bkh, g);
      float sqj = sqBH[j0 + t2 * 16 + n];
#pragma unroll
      for (int i = 0; i < 8; ++i) {
        float d2 = fmaxf(sqi[i] + sqj - 2.0f * g[i], 0.0f);
        sv[t2][i] = CL * s[i] - CG * d2;
      }
    }

    float arow[8];
#pragma unroll
    for (int i = 0; i < 8; ++i) {
      float sa = sv[0][i], sb = sv[1][i];
      // row lives in one 16-lane half: xor-reduce over lanes 1,2,4,8
      float mt = fmaxf(sa, sb);
#pragma unroll
      for (int msk = 1; msk < 16; msk <<= 1) mt = fmaxf(mt, __shfl_xor(mt, msk, 32));
      float mnew = fmaxf(mrow[i], mt);
      float alpha = __expf(mrow[i] - mnew);
      float p0 = __expf(sa - mnew);
      float p1 = __expf(sb - mnew);
      float ps = p0 + p1;
#pragma unroll
      for (int msk = 1; msk < 16; msk <<= 1) ps += __shfl_xor(ps, msk, 32);
      lrow[i] = lrow[i] * alpha + ps;
      mrow[i] = mnew;
      arow[i] = alpha;
      int il = half * 8 + i;                 // query-local row
      pLds[il * 32 + n]      = (bf16_t)p0;   // key-local col n
      pLds[il * 32 + 16 + n] = (bf16_t)p1;   // key-local col 16+n
    }
#pragma unroll
    for (int t = 0; t < 4; ++t)
#pragma unroll
      for (int i = 0; i < 8; ++i) o[t][i] *= arow[i];
    __syncthreads(); // make P visible (and ordered) in LDS

    v16bf ap = load_a_frag((const bf16_t*)pLds, 32, lane); // P as 16x32 A-frag
    const size_t vBase = ((size_t)(b * NHEAD + h) * DHEAD) * NTOK + j0;
#pragma unroll
    for (int t = 0; t < 4; ++t) {
      v16bf bv = load_b_frag(Vt + vBase + (size_t)(t * 16) * NTOK, NTOK, lane);
      o[t] = wmma_bf16(ap, bv, o[t]);
    }
    __syncthreads();
  }

#pragma unroll
  for (int i = 0; i < 8; ++i) {
    float inv = 1.0f / lrow[i];
#pragma unroll
    for (int t = 0; t < 4; ++t) o[t][i] *= inv;
  }
#pragma unroll
  for (int t = 0; t < 4; ++t)
#pragma unroll
    for (int i = 0; i < 8; ++i) {
      int row = b * NTOK + qrow0 + half * 8 + i;
      O[(size_t)row * D_MODEL + h * DHEAD + t * 16 + n] = (bf16_t)o[t][i];
    }
}

// ---------------------------------------------------------------------------
// Launch: convert -> project (Q,K,Vt) -> norms -> fused attention -> out proj
// Workspace map (bytes):
//   [0,8M)   x bf16            [8M,10M)  WqT   [10M,12M) WkT
//   [12M,14M) WvT              [14M,16M) WoT
//   [16M,24M) Q bf16           [24M,32M) K bf16
//   [32M,40M) Vt bf16          [40M,48M) O bf16
//   [48M,48.25M) key sq-norms f32          total ~48.3 MB
// ---------------------------------------------------------------------------
extern "C" void kernel_launch(void* const* d_in, const int* in_sizes, int n_in,
                              void* d_out, int out_size, void* d_ws, size_t ws_size,
                              hipStream_t stream) {
  const float* x   = (const float*)d_in[0];
  const float* w_q = (const float*)d_in[1];
  const float* w_k = (const float*)d_in[2];
  const float* w_v = (const float*)d_in[3];
  const float* w_o = (const float*)d_in[4];
  (void)in_sizes; (void)n_in; (void)out_size; (void)ws_size;

  char* ws = (char*)d_ws;
  const size_t MB = (size_t)1 << 20;
  bf16_t* XB  = (bf16_t*)(ws + 0);
  bf16_t* WQT = (bf16_t*)(ws + 8 * MB);
  bf16_t* WKT = (bf16_t*)(ws + 10 * MB);
  bf16_t* WVT = (bf16_t*)(ws + 12 * MB);
  bf16_t* WOT = (bf16_t*)(ws + 14 * MB);
  bf16_t* Qb  = (bf16_t*)(ws + 16 * MB);
  bf16_t* Kb  = (bf16_t*)(ws + 24 * MB);
  bf16_t* VT  = (bf16_t*)(ws + 32 * MB);
  bf16_t* Ob  = (bf16_t*)(ws + 40 * MB);
  float*  SQ  = (float*) (ws + 48 * MB);

  const int nx = BATCH * NTOK * D_MODEL; // 4,194,304
  cvt_f32_bf16<<<4096, 256, 0, stream>>>(x, XB, nx);
  transpose_w_bf16<<<4096, 256, 0, stream>>>(w_q, WQT);
  transpose_w_bf16<<<4096, 256, 0, stream>>>(w_k, WKT);
  transpose_w_bf16<<<4096, 256, 0, stream>>>(w_v, WVT);
  transpose_w_bf16<<<4096, 256, 0, stream>>>(w_o, WOT);

  dim3 gg(MTOT / 64, D_MODEL / 64);
  gemm_proj<0><<<gg, 128, 0, stream>>>(XB, WQT, (void*)Qb);
  gemm_proj<0><<<gg, 128, 0, stream>>>(XB, WKT, (void*)Kb);
  gemm_proj<1><<<gg, 128, 0, stream>>>(XB, WVT, (void*)VT);

  k_sqnorm<<<(BATCH * NHEAD * NTOK) / 256, 256, 0, stream>>>(Kb, SQ);

  attn_kernel<<<dim3(NTOK / 16, NHEAD, BATCH), 32, 0, stream>>>(Qb, Kb, VT, SQ, Ob);

  gemm_proj<2><<<gg, 128, 0, stream>>>(Ob, WOT, d_out);
}
